// GCN_23914377904381
// MI455X (gfx1250) — matrix operand order repaired and checked by reading, not compiled, not run on hardware
//
#include <hip/hip_runtime.h>
#include <hip/hip_bf16.h>

typedef __attribute__((ext_vector_type(16))) __bf16 v16bf;
typedef __attribute__((ext_vector_type(8)))  float  v8f;
typedef __attribute__((ext_vector_type(4)))  int    v4i;

#if defined(__has_builtin)
#if __has_builtin(__builtin_amdgcn_global_load_async_to_lds_b128)
#define HAVE_ASYNC_LDS 1
#endif
#if __has_builtin(__builtin_amdgcn_s_wait_asynccnt)
#define HAVE_WAIT_ASYNC 1
#endif
#endif

typedef __attribute__((address_space(1))) v4i* gvec_t;   // global int4*
typedef __attribute__((address_space(3))) v4i* lvec_t;   // LDS int4*

// ---------------- utility kernels ----------------

__global__ void zero_f32(float* __restrict__ p, int n) {
  int i = blockIdx.x * blockDim.x + threadIdx.x;
  if (i < n) p[i] = 0.0f;
}

__global__ void cvt_bf16(const float* __restrict__ x, __bf16* __restrict__ y, int n) {
  int i = blockIdx.x * blockDim.x + threadIdx.x;
  if (i < n) y[i] = (__bf16)x[i];
}

__global__ void degree_kernel(const int* __restrict__ src, const int* __restrict__ dst,
                              float* __restrict__ outdeg, float* __restrict__ indeg, int nE) {
  int e = blockIdx.x * blockDim.x + threadIdx.x;
  if (e < nE) {
    atomicAdd(&outdeg[src[e]], 1.0f);
    atomicAdd(&indeg[dst[e]], 1.0f);
  }
}

__global__ void norm_kernel(const float* __restrict__ outdeg, const float* __restrict__ indeg,
                            float* __restrict__ norm_out, float* __restrict__ norm_in, int n) {
  int i = blockIdx.x * blockDim.x + threadIdx.x;
  if (i < n) {
    norm_out[i] = rsqrtf(fmaxf(outdeg[i], 1.0f));
    norm_in[i]  = rsqrtf(fmaxf(indeg[i], 1.0f));
  }
}

// Pack W[k][n] (K=128 x ncols) into WMMA B-fragment order (bf16).
// 16-bit B 32x16 layout (ISA 7.12.2): lane L -> col n=L&15; lanes 0-15 K 0..7,16..23,
// lanes 16-31 K 8..15,24..31, 2 K-values per VGPR.
__global__ void pack_weights(const float* __restrict__ W, __bf16* __restrict__ pack, int ncols) {
  int idx = blockIdx.x * blockDim.x + threadIdx.x;
  int NT = ncols >> 4;
  int total = 4 * NT * 512;
  if (idx >= total) return;
  int j  = idx & 15;
  int L  = (idx >> 4) & 31;
  int nt = (idx >> 9) % NT;
  int kt = (idx >> 9) / NT;
  int k  = kt * 32 + (L >> 4) * 8 + (j & 7) + (j >> 3) * 16;
  int n  = nt * 16 + (L & 15);
  pack[idx] = (__bf16)W[k * ncols + n];
}

// ---------------- WMMA GEMM: Y[nrows x NCOLS] = A_bf16[nrows x 128] @ Wpack ----------------
// One block = one 16-row node tile; wave w computes column tile w.
template <int NCOLS>
__global__ __launch_bounds__(32 * (NCOLS / 16))
void gemm_wmma(const __bf16* __restrict__ A, const __bf16* __restrict__ wpack,
               float* __restrict__ Y, int nrows) {
  constexpr int NT = NCOLS / 16;
  constexpr int NTHREADS = 32 * NT;
  __shared__ __bf16 ldsA[16 * 128];                 // 4 KB tile

  const int tid  = threadIdx.x;
  const int row0 = blockIdx.x * 16;
  const bool full = (row0 + 16 <= nrows);
  const __bf16* arow = A + (size_t)row0 * 128;

#if HAVE_ASYNC_LDS
  if (full) {
    // 4096 bytes = 256 x b128 async transfers straight into LDS (ASYNCcnt path).
    for (int i = tid; i < 256; i += NTHREADS) {
      char* gsrc = (char*)arow + (size_t)i * 16;    // drop const in generic AS
      __builtin_amdgcn_global_load_async_to_lds_b128(
          (gvec_t)gsrc,
          (lvec_t)((__attribute__((address_space(3))) char*)ldsA + i * 16),
          0, 0);
    }
#if HAVE_WAIT_ASYNC
    __builtin_amdgcn_s_wait_asynccnt(0);
#else
    asm volatile("s_wait_asynccnt 0" ::: "memory");
#endif
  } else
#endif
  {
    for (int i = tid; i < 16 * 128; i += NTHREADS) {
      int r = i >> 7;
      ldsA[i] = (row0 + r < nrows) ? arow[i] : (__bf16)0.0f;
    }
  }
  __syncthreads();

  const int wave = tid >> 5;     // == column tile nt
  const int lane = tid & 31;
  const int m  = lane & 15;
  const int kh = lane >> 4;

  v8f c = {};
  const v16bf* wb = (const v16bf*)wpack;

#pragma unroll
  for (int kt = 0; kt < 4; ++kt) {
    // A fragment (16x32 bf16): two contiguous 16B chunks per lane from LDS.
    v16bf a;
    const __bf16* ar = &ldsA[m * 128 + kt * 32 + kh * 8];
#pragma unroll
    for (int v = 0; v < 4; ++v) {
      a[2 * v]         = ar[2 * v];
      a[2 * v + 1]     = ar[2 * v + 1];
      a[8 + 2 * v]     = ar[16 + 2 * v];
      a[8 + 2 * v + 1] = ar[16 + 2 * v + 1];
    }
    // B fragment: one contiguous 32B load from packed weights.
    v16bf b = wb[(kt * NT + wave) * 32 + lane];
    c = __builtin_amdgcn_wmma_f32_16x16x32_bf16(false, a, false, b, (short)0, c,
                                                false, false);
  }

  // C/D layout: lane L -> col L&15; VGPR v -> row v + 8*(L>>4).
  float* yout = Y + (size_t)row0 * NCOLS + wave * 16 + (lane & 15);
  if (full) {
#pragma unroll
    for (int v = 0; v < 8; ++v)
      yout[(size_t)(v + 8 * kh) * NCOLS] = c[v];
  } else {
#pragma unroll
    for (int v = 0; v < 8; ++v) {
      int r = v + 8 * kh;
      if (row0 + r < nrows) yout[(size_t)r * NCOLS] = c[v];
    }
  }
}

// ---------------- SpMM scatter: agg[dst] += norm_out[src] * Y[src] ----------------
template <int LOGF>
__global__ void spmm_scatter(const float* __restrict__ Y,
                             const int* __restrict__ src, const int* __restrict__ dst,
                             const float* __restrict__ norm_out,
                             float* __restrict__ agg, int nE) {
  constexpr int F = 1 << LOGF;
  long long t = (long long)blockIdx.x * blockDim.x + threadIdx.x;
  int e = (int)(t >> LOGF);
  int f = (int)t & (F - 1);
  if (e < nE) {
    int s = src[e];
    int d = dst[e];
    float v = Y[(size_t)s * F + f] * norm_out[s];
    atomicAdd(&agg[(size_t)d * F + f], v);
  }
}

// ---------------- out = relu(agg * norm_in + b), templated output type ----------------
template <int LOGF, typename OutT>
__global__ void bias_relu(const float* __restrict__ agg, const float* __restrict__ norm_in,
                          const float* __restrict__ b, OutT* __restrict__ out, int n_nodes) {
  constexpr int F = 1 << LOGF;
  long long t = (long long)blockIdx.x * blockDim.x + threadIdx.x;
  if (t < ((long long)n_nodes << LOGF)) {
    int n = (int)(t >> LOGF);
    int f = (int)t & (F - 1);
    out[t] = (OutT)fmaxf(agg[t] * norm_in[n] + b[f], 0.0f);
  }
}

// ---------------- launch ----------------
extern "C" void kernel_launch(void* const* d_in, const int* in_sizes, int n_in,
                              void* d_out, int out_size, void* d_ws, size_t ws_size,
                              hipStream_t stream) {
  const float* h  = (const float*)d_in[0];
  const float* W1 = (const float*)d_in[1];
  const float* b1 = (const float*)d_in[2];
  const float* W2 = (const float*)d_in[3];
  const float* b2 = (const float*)d_in[4];
  const int*   src = (const int*)d_in[5];
  const int*   dst = (const int*)d_in[6];

  const int IN  = 128;
  const int N   = in_sizes[0] / IN;         // 50000
  const int nE  = in_sizes[5];              // 800000
  (void)ws_size; (void)n_in; (void)out_size;

  // Workspace layout (256B-aligned slices).
  char* ws = (char*)d_ws;
  size_t off = 0;
  auto carve = [&](size_t bytes) -> char* {
    char* p = ws + off;
    off += (bytes + 255) & ~(size_t)255;
    return p;
  };
  float*  agg1     = (float*)carve((size_t)N * 128 * 4);
  float*  y1       = (float*)carve((size_t)N * 128 * 4);
  float*  agg2     = (float*)carve((size_t)N * 64 * 4);
  float*  y2       = (float*)carve((size_t)N * 64 * 4);
  __bf16* abf      = (__bf16*)carve((size_t)N * 128 * 2);  // bf16(h), then bf16(h1)
  float*  outdeg   = (float*)carve((size_t)N * 4);
  float*  indeg    = (float*)carve((size_t)N * 4);
  float*  norm_out = (float*)carve((size_t)N * 4);
  float*  norm_in  = (float*)carve((size_t)N * 4);
  __bf16* w1pack   = (__bf16*)carve((size_t)128 * 128 * 2);
  __bf16* w2pack   = (__bf16*)carve((size_t)128 * 64 * 2);

  const int TB = 256;
  auto blk = [&](long long n) { return (unsigned)((n + TB - 1) / TB); };

  // 0) zero accumulators
  zero_f32<<<blk((long long)N * 128), TB, 0, stream>>>(agg1, N * 128);
  zero_f32<<<blk((long long)N * 64),  TB, 0, stream>>>(agg2, N * 64);
  zero_f32<<<blk(N), TB, 0, stream>>>(outdeg, N);
  zero_f32<<<blk(N), TB, 0, stream>>>(indeg, N);

  // 1) degrees + norms
  degree_kernel<<<blk(nE), TB, 0, stream>>>(src, dst, outdeg, indeg, nE);
  norm_kernel<<<blk(N), TB, 0, stream>>>(outdeg, indeg, norm_out, norm_in, N);

  // 2) weights -> packed bf16 fragments; h -> bf16
  pack_weights<<<blk(4 * 8 * 512), TB, 0, stream>>>(W1, w1pack, 128);
  pack_weights<<<blk(4 * 4 * 512), TB, 0, stream>>>(W2, w2pack, 64);
  cvt_bf16<<<blk((long long)N * 128), TB, 0, stream>>>(h, abf, N * 128);

  const unsigned ntiles = (unsigned)((N + 15) / 16);

  // 3) layer 1: y1 = h @ W1 (WMMA, async LDS staging), 128-wide SpMM, bias+relu -> bf16 h1
  gemm_wmma<128><<<ntiles, 256, 0, stream>>>(abf, w1pack, y1, N);
  spmm_scatter<7><<<blk((long long)nE << 7), TB, 0, stream>>>(y1, src, dst, norm_out, agg1, nE);
  bias_relu<7, __bf16><<<blk((long long)N << 7), TB, 0, stream>>>(agg1, norm_in, b1, abf, N);

  // 4) layer 2 (GEMM-before-SpMM halves scatter width to 64)
  gemm_wmma<64><<<ntiles, 128, 0, stream>>>(abf, w2pack, y2, N);
  spmm_scatter<6><<<blk((long long)nE << 6), TB, 0, stream>>>(y2, src, dst, norm_out, agg2, nE);
  bias_relu<6, float><<<blk((long long)N << 6), TB, 0, stream>>>(agg2, norm_in, b2, (float*)d_out, N);
}
